// VectorQuantizer_12421045420169
// MI455X (gfx1250) — matrix operand order repaired and checked by reading, not compiled, not run on hardware
//
#include <hip/hip_runtime.h>
#include <hip/hip_fp16.h>

typedef __attribute__((ext_vector_type(16))) _Float16 v16h;
typedef __attribute__((ext_vector_type(8)))  _Float16 v8h;
typedef __attribute__((ext_vector_type(8)))  float    v8f;
typedef __attribute__((ext_vector_type(4)))  float    v4f;

#define NB    64
#define ND    64          // C == dim
#define NHW   1024        // H*W
#define NTOK  65536       // NB*NHW
#define NK    1024        // codebook size
#define NELEM 4194304     // NB*ND*NHW
#define BETA  0.25f

// ---------------- ws layout (bytes) ----------------
// [0,       131072)  cb16     : NK*ND f16  (128 KB)
// [131072,  135168)  halfnorm : NK   f32   (  4 KB)
// [135168,  397312)  idx      : NTOK i32   (256 KB)
// [397312,  401408)  hist     : NK   i32   (  4 KB)
// [401408,  466944)  partials : 16384 f32  ( 64 KB)

// ---- prep: f16 codebook, half-norms, zero histogram ----
__global__ __launch_bounds__(256)
void vq_prep(const float* __restrict__ cb, _Float16* __restrict__ cb16,
             float* __restrict__ hn, int* __restrict__ hist) {
    int k = blockIdx.x * 256 + threadIdx.x;          // 0..1023 (grid = 4 blocks)
    float s = 0.0f;
    #pragma unroll 8
    for (int d = 0; d < ND; ++d) {
        float v = cb[k * ND + d];
        s += v * v;
        cb16[k * ND + d] = (_Float16)v;
    }
    hn[k] = 0.5f * s;
    hist[k] = 0;
}

// ---- main: LDS-staged WMMA score GEMM + argmax -> idx, histogram ----
// 512 workgroups x 256 threads; each WG owns 128 tokens of one image.
// Codebook swept in 4 phases of 256 codes staged through LDS (B fed by ds_load).
__global__ __launch_bounds__(256)
void vq_argmin(const float* __restrict__ z, const _Float16* __restrict__ cb16,
               const float* __restrict__ hn, int* __restrict__ idx,
               int* __restrict__ hist) {
    __shared__ _Float16 lds_z[128 * 72];             // 18 KB: 128 tokens x 64 dims (padded)
    __shared__ _Float16 lds_cb[256 * 72];            // 36 KB: 256 codes x 64 dims (padded)
    __shared__ float    lds_hn[256];                 //  1 KB

    const int tid = threadIdx.x;
    const int wg  = blockIdx.x;
    const int b   = wg >> 3;                         // (wg*128)/1024
    const int hw0 = (wg & 7) << 7;                   // (wg*128)%1024

    // stage z tile to LDS as f16 (coalesced: contiguous over tokens per dim)
    const float* zb = z + (size_t)b * (size_t)(ND * NHW) + hw0;
    for (int i = tid; i < 128 * ND; i += 256) {
        int d = i >> 7, t = i & 127;
        lds_z[t * 72 + d] = (_Float16)zb[d * NHW + t];
    }
    __syncthreads();

    const int wave = tid >> 5;
    const int lane = tid & 31;
    const int col  = lane & 15;                      // M (for A) / N (for B/D)
    const int half = lane >> 4;                      // K-half selector

    // A operands (held in VGPRs for the whole sweep): wave's 16 tokens, K 0..63
    const _Float16* zr = &lds_z[(wave * 16 + col) * 72];
    v8h p0 = *(const v8h*)(zr +      half * 8);
    v8h p1 = *(const v8h*)(zr + 16 + half * 8);
    v8h p2 = *(const v8h*)(zr + 32 + half * 8);
    v8h p3 = *(const v8h*)(zr + 48 + half * 8);
    v16h a0 = __builtin_shufflevector(p0, p1, 0,1,2,3,4,5,6,7,8,9,10,11,12,13,14,15);
    v16h a1 = __builtin_shufflevector(p2, p3, 0,1,2,3,4,5,6,7,8,9,10,11,12,13,14,15);

    float best[8];
    int   bidx[8];
    #pragma unroll
    for (int v = 0; v < 8; ++v) { best[v] = -3.4e38f; bidx[v] = 0; }

    for (int p = 0; p < 4; ++p) {
        // stage 256 codes (f16, padded rows) + their half-norms into LDS
        const _Float16* cg = cb16 + (p << 8) * ND;
        #pragma unroll
        for (int j = 0; j < 8; ++j) {                // 2048 16B-units / 256 threads
            int u = tid + j * 256;
            int k = u >> 3, doff = (u & 7) * 8;
            *(v8h*)(lds_cb + k * 72 + doff) = *(const v8h*)(cg + k * ND + doff);
        }
        if (tid < 256) lds_hn[tid] = hn[(p << 8) + tid];
        __syncthreads();

        // score = z.c - 0.5*||c||^2 ; argmax over codes
        for (int kt = 0; kt < 16; ++kt) {
            const int nl = kt * 16 + col;            // code within staged chunk
            const _Float16* cr = lds_cb + nl * 72 + half * 8;
            v8h q0 = *(const v8h*)(cr);
            v8h q1 = *(const v8h*)(cr + 16);
            v8h q2 = *(const v8h*)(cr + 32);
            v8h q3 = *(const v8h*)(cr + 48);
            v16h b0 = __builtin_shufflevector(q0, q1, 0,1,2,3,4,5,6,7,8,9,10,11,12,13,14,15);
            v16h b1 = __builtin_shufflevector(q2, q3, 0,1,2,3,4,5,6,7,8,9,10,11,12,13,14,15);
            v8f acc = {};
            acc = __builtin_amdgcn_wmma_f32_16x16x32_f16(false, a0, false, b0,
                                                         (short)0, acc, false, false);
            acc = __builtin_amdgcn_wmma_f32_16x16x32_f16(false, a1, false, b1,
                                                         (short)0, acc, false, false);
            const float h = lds_hn[nl];
            const int   n = (p << 8) + nl;
            #pragma unroll
            for (int v = 0; v < 8; ++v) {
                float s = acc[v] - h;
                if (s > best[v]) { best[v] = s; bidx[v] = n; }
            }
        }
        __syncthreads();
    }

    // reduce argmax across the 16 columns (xor stays inside each 16-lane half)
    #pragma unroll
    for (int v = 0; v < 8; ++v) {
        #pragma unroll
        for (int off = 1; off < 16; off <<= 1) {
            float ob = __shfl_xor(best[v], off, 32);
            int   oi = __shfl_xor(bidx[v], off, 32);
            if (ob > best[v] || (ob == best[v] && oi < bidx[v])) {
                best[v] = ob; bidx[v] = oi;
            }
        }
    }
    if (col == 0) {
        int tbase = wg * 128 + wave * 16 + half * 8; // D row M = v + 8*half
        #pragma unroll
        for (int v = 0; v < 8; ++v) {
            idx[tbase + v] = bidx[v];
            atomicAdd(&hist[bidx[v]], 1);            // integer: order-independent
        }
    }
}

// ---- gather/output/loss partials (memory-bound, L2-resident) ----
__global__ __launch_bounds__(256)
void vq_out(const float* __restrict__ z, const float* __restrict__ cb,
            const int* __restrict__ idx, float* __restrict__ out,
            float* __restrict__ partials) {
    __shared__ float wsum[8];
    int i = blockIdx.x * 256 + threadIdx.x;          // element in NCHW order
    int d   = (i >> 10) & 63;
    int tok = ((i >> 16) << 10) | (i & 1023);        // b*1024 + hw
    int k = idx[tok];
    float q  = cb[k * ND + d];
    float zv = z[i];
    out[i] = q;                                      // straight-through fwd == z_q
    float df = q - zv;
    float s = df * df;
    #pragma unroll
    for (int off = 16; off >= 1; off >>= 1) s += __shfl_xor(s, off, 32);
    if ((threadIdx.x & 31) == 0) wsum[threadIdx.x >> 5] = s;
    __syncthreads();
    if (threadIdx.x == 0) {
        float t = 0.0f;
        #pragma unroll
        for (int w = 0; w < 8; ++w) t += wsum[w];    // fixed order: deterministic
        partials[blockIdx.x] = t;
    }
}

// ---- finalize: loss + perplexity (single block, deterministic trees) ----
__global__ __launch_bounds__(1024)
void vq_final(const float* __restrict__ partials, const int* __restrict__ hist,
              const int* __restrict__ flg, float* __restrict__ out) {
    __shared__ float red[1024];
    const int t = threadIdx.x;
    float s = 0.0f;
    #pragma unroll
    for (int j = 0; j < 16; ++j) s += partials[t + j * 1024];
    red[t] = s;
    __syncthreads();
    for (int off = 512; off >= 1; off >>= 1) {
        if (t < off) red[t] += red[t + off];
        __syncthreads();
    }
    float sumsq = red[0];
    __syncthreads();

    float e = (float)hist[t] * (1.0f / (float)NTOK);
    red[t] = e * logf(e + 1e-10f);
    __syncthreads();
    for (int off = 512; off >= 1; off >>= 1) {
        if (t < off) red[t] += red[t + off];
        __syncthreads();
    }
    if (t == 0) {
        out[NELEM]     = flg[0] ? (1.0f + BETA) * sumsq * (1.0f / (float)NELEM) : 0.0f;
        out[NELEM + 1] = expf(-red[0]);
    }
}

extern "C" void kernel_launch(void* const* d_in, const int* in_sizes, int n_in,
                              void* d_out, int out_size, void* d_ws, size_t ws_size,
                              hipStream_t stream) {
    const float* z   = (const float*)d_in[0];
    const float* cb  = (const float*)d_in[1];
    // d_in[2] (codebook_weight) is unused by the reference
    const int*   flg = (const int*)d_in[3];
    float* out = (float*)d_out;

    char* ws = (char*)d_ws;
    _Float16* cb16     = (_Float16*)(ws);
    float*    hn       = (float*)(ws + 131072);
    int*      idx      = (int*)(ws + 135168);
    int*      hist     = (int*)(ws + 397312);
    float*    partials = (float*)(ws + 401408);

    vq_prep  <<<4,            256,  0, stream>>>(cb, cb16, hn, hist);
    vq_argmin<<<NTOK / 128,   256,  0, stream>>>(z, cb16, hn, idx, hist);
    vq_out   <<<NELEM / 256,  256,  0, stream>>>(z, cb, idx, out, partials);
    vq_final <<<1,            1024, 0, stream>>>(partials, hist, flg, out);
}
